// GraphNeuralPDESolver_77945066488623
// MI455X (gfx1250) — matrix operand order repaired and checked by reading, not compiled, not run on hardware
//
#include <hip/hip_runtime.h>

// ---------------------------------------------------------------------------
// CDNA5 (gfx1250) WMMA types
// ---------------------------------------------------------------------------
typedef __bf16 bf16_t;
typedef bf16_t v16bf __attribute__((ext_vector_type(16)));
typedef float  v8f   __attribute__((ext_vector_type(8)));

#define BB 8  // BATCH

// activation: 0 = identity, 1 = swish (x * sigmoid(x)) via fast rcp
template <int ACT>
__device__ __forceinline__ float apply_act(float x)
{
    if (ACT == 1) {
        const float s = __builtin_amdgcn_rcpf(1.f + __expf(-x));
        return x * s;
    }
    return x;
}

// ---------------------------------------------------------------------------
// Weight pack: W[K,128] f32 -> B-fragment bf16 layout
//   Wp[ kbg*4096 + col*32 + h*16 + e ] = W[(kbg*32 + 16h + e)*128 + col]
// ---------------------------------------------------------------------------
__global__ void pack_w_k(const float* __restrict__ W, bf16_t* __restrict__ Wp, int K)
{
    const int t = blockIdx.x * blockDim.x + threadIdx.x;
    const int total = (K >> 5) * 4096;
    if (t >= total) return;
    const int e   = t & 15;
    const int h   = (t >> 4) & 1;
    const int col = (t >> 5) & 127;
    const int kbg = t >> 12;
    const int k   = kbg * 32 + 16 * h + e;
    Wp[t] = (bf16_t)W[(size_t)k * 128 + col];
}

// ---------------------------------------------------------------------------
// FAST path: out[M,128] = act( A[M, NSEG*128] @ W + b )
// A = virtual concat of NSEG 128-wide sources; mode 0 direct, 1 gather.
// Block: 256 thr = 8 waves; tile 16 rows x 128 cols; 1 WMMA tile per wave.
// ---------------------------------------------------------------------------
template <int NSEG, int ACT>
__global__ __launch_bounds__(256) void wmma_mlp_fast(
    const float* __restrict__ p0, const int* __restrict__ x0, int m0,
    const float* __restrict__ p1, const int* __restrict__ x1, int m1,
    const float* __restrict__ p2, const int* __restrict__ x2, int m2,
    const bf16_t* __restrict__ Wp, const float* __restrict__ bias,
    float* __restrict__ out, int M)
{
    const int lane = threadIdx.x & 31;
    const int wave = threadIdx.x >> 5;
    const int h    = lane >> 4;
    const int l16  = lane & 15;
    const int row0 = blockIdx.x * 16;
    const int colW = wave * 16 + l16;

    v8f acc;
    const float bv = bias[colW];
#pragma unroll
    for (int i = 0; i < 8; ++i) acc[i] = bv;

    int rowA = row0 + l16;
    if (rowA >= M) rowA = M - 1;      // clamp; OOB rows never stored
    const int ent = rowA >> 3, bb = rowA & 7;

    const float* sp[3] = {p0, p1, p2};
    const int*   sx[3] = {x0, x1, x2};
    const int    sm[3] = {m0, m1, m2};

#pragma unroll
    for (int seg = 0; seg < NSEG; ++seg) {
        size_t srow;
        if (sm[seg] == 0) srow = (size_t)rowA;
        else              srow = (size_t)sx[seg][ent] * BB + bb;
        const float* arow = sp[seg] + srow * 128;
#pragma unroll
        for (int kb = 0; kb < 4; ++kb) {
            const int kk = kb * 32;
            // A frag: elem e -> k=kk+16h+e
            const float4 a0 = *(const float4*)(arow + kk + 16 * h);
            const float4 a1 = *(const float4*)(arow + kk + 16 * h + 4);
            const float4 a2 = *(const float4*)(arow + kk + 16 * h + 8);
            const float4 a3 = *(const float4*)(arow + kk + 16 * h + 12);
            v16bf a;
            a[0]  = (bf16_t)a0.x; a[1]  = (bf16_t)a0.y; a[2]  = (bf16_t)a0.z; a[3]  = (bf16_t)a0.w;
            a[4]  = (bf16_t)a1.x; a[5]  = (bf16_t)a1.y; a[6]  = (bf16_t)a1.z; a[7]  = (bf16_t)a1.w;
            a[8]  = (bf16_t)a2.x; a[9]  = (bf16_t)a2.y; a[10] = (bf16_t)a2.z; a[11] = (bf16_t)a2.w;
            a[12] = (bf16_t)a3.x; a[13] = (bf16_t)a3.y; a[14] = (bf16_t)a3.z; a[15] = (bf16_t)a3.w;
            // B frag: prepacked contiguous 16 bf16 for this (k-block, col, h)
            const int kbg = seg * 4 + kb;
            const v16bf b = *(const v16bf*)(Wp + (((size_t)kbg * 128 + colW) * 2 + h) * 16);
            acc = __builtin_amdgcn_wmma_f32_16x16x32_bf16(
                false, a, false, b, (short)0, acc, false, false);
        }
    }

    // epilogue: C/D lane holds col colW; VGPR r holds row row0 + r + 8h.
    float* base = out + (size_t)(row0 + 8 * h) * 128 + colW;
    if (row0 + 16 <= M) {
#pragma unroll
        for (int r = 0; r < 8; ++r)
            base[(size_t)r * 128] = apply_act<ACT>(acc[r]);
    } else {
#pragma unroll
        for (int r = 0; r < 8; ++r) {
            const int row = row0 + r + 8 * h;
            if (row < M) base[(size_t)r * 128] = apply_act<ACT>(acc[r]);
        }
    }
}

// ---------------------------------------------------------------------------
// GENERIC path (embedder first layers only, K <= 7, one WMMA, swish).
// Up to 2 sources; mode 0 direct, 2 broadcast (entity-level, no batch).
// ---------------------------------------------------------------------------
__global__ __launch_bounds__(256) void wmma_mlp_generic(
    const float* __restrict__ s0p, int s0d, int s0m,
    const float* __restrict__ s1p, int s1d, int s1m,
    const float* __restrict__ Wt, const float* __restrict__ bias,
    float* __restrict__ out, int M, int K)
{
    const int lane = threadIdx.x & 31;
    const int wave = threadIdx.x >> 5;
    const int h    = lane >> 4;
    const int l16  = lane & 15;
    const int row0 = blockIdx.x * 16;
    const int colW = wave * 16 + l16;

    v8f acc;
    const float bv = bias[colW];
#pragma unroll
    for (int i = 0; i < 8; ++i) acc[i] = bv;

    int rowA = row0 + l16;
    if (rowA >= M) rowA = M - 1;
    const int ent = rowA >> 3;

    v16bf a, b;
#pragma unroll
    for (int e = 0; e < 16; ++e) {
        const int k = 16 * h + e;
        float v = 0.f;
        if (k < K) {
            if (k < s0d) {
                const size_t srow = (s0m == 0) ? (size_t)rowA : (size_t)ent;
                v = s0p[srow * (size_t)s0d + k];
            } else {
                const int kk = k - s0d;
                const size_t srow = (s1m == 0) ? (size_t)rowA : (size_t)ent;
                v = s1p[srow * (size_t)s1d + kk];
            }
        }
        a[e] = (bf16_t)v;
    }
#pragma unroll
    for (int e = 0; e < 16; ++e) {
        const int k = 16 * h + e;
        b[e] = (bf16_t)((k < K) ? Wt[(size_t)k * 128 + colW] : 0.f);
    }
    acc = __builtin_amdgcn_wmma_f32_16x16x32_bf16(
        false, a, false, b, (short)0, acc, false, false);

    float* base = out + (size_t)(row0 + 8 * h) * 128 + colW;
    if (row0 + 16 <= M) {
#pragma unroll
        for (int r = 0; r < 8; ++r)
            base[(size_t)r * 128] = apply_act<1>(acc[r]);
    } else {
#pragma unroll
        for (int r = 0; r < 8; ++r) {
            const int row = row0 + r + 8 * h;
            if (row < M) base[(size_t)r * 128] = apply_act<1>(acc[r]);
        }
    }
}

// ---------------------------------------------------------------------------
// LayerNorm(128) + scale/shift + optional residual. One wave per row.
// ---------------------------------------------------------------------------
__global__ __launch_bounds__(256) void ln_res_k(
    const float* __restrict__ lin, const float* __restrict__ s,
    const float* __restrict__ bsh, const float* __restrict__ res,
    float* __restrict__ dst, int M)
{
    const int wid  = (int)((blockIdx.x * (size_t)blockDim.x + threadIdx.x) >> 5);
    const int lane = threadIdx.x & 31;
    if (wid >= M) return;
    const size_t base = (size_t)wid * 128;
    float v0 = lin[base + lane], v1 = lin[base + lane + 32];
    float v2 = lin[base + lane + 64], v3 = lin[base + lane + 96];
    float sum = v0 + v1 + v2 + v3;
#pragma unroll
    for (int m = 16; m >= 1; m >>= 1) sum += __shfl_xor(sum, m, 32);
    const float mu = sum * (1.f / 128.f);
    float d0 = v0 - mu, d1 = v1 - mu, d2 = v2 - mu, d3 = v3 - mu;
    float var = d0 * d0 + d1 * d1 + d2 * d2 + d3 * d3;
#pragma unroll
    for (int m = 16; m >= 1; m >>= 1) var += __shfl_xor(var, m, 32);
    const float inv = rsqrtf(var * (1.f / 128.f) + 1e-5f);
    float dv[4] = {d0, d1, d2, d3};
#pragma unroll
    for (int j = 0; j < 4; ++j) {
        const int c = lane + 32 * j;
        float y = dv[j] * inv * s[c] + bsh[c];
        if (res) y += res[base + c];
        dst[base + c] = y;
    }
}

// ---------------------------------------------------------------------------
// Segment mean helpers (atomic sum + count, then divide)
// ---------------------------------------------------------------------------
__global__ void scatter_add_k(const float* __restrict__ feat,
                              const int* __restrict__ idx,
                              float* __restrict__ agg, long total)
{
    const long t = (long)blockIdx.x * blockDim.x + threadIdx.x;
    if (t >= total) return;
    const int  c  = (int)(t & 127);
    const long rb = t >> 7;
    const int  b  = (int)(rb & 7);
    const long e  = rb >> 3;
    atomicAdd(&agg[((size_t)idx[e] * BB + b) * 128 + c], feat[t]);
}

__global__ void count_k(const int* __restrict__ idx, float* __restrict__ cnt, int E)
{
    const int t = blockIdx.x * blockDim.x + threadIdx.x;
    if (t < E) atomicAdd(&cnt[idx[t]], 1.0f);
}

__global__ void divide_k(float* __restrict__ agg, const float* __restrict__ cnt, long rows)
{
    const long t = (long)blockIdx.x * blockDim.x + threadIdx.x;
    if (t >= rows * 128) return;
    const long ent = (t >> 7) >> 3;
    agg[t] /= fmaxf(cnt[ent], 1.0f);
}

// ---------------------------------------------------------------------------
// Final 128 -> 3 layer + residual: out = u + h2 @ W2 + b2
// ---------------------------------------------------------------------------
__global__ void out_final_k(const float* __restrict__ h2, const float* __restrict__ W,
                            const float* __restrict__ bias, const float* __restrict__ u,
                            float* __restrict__ out, int M)
{
    const int t = blockIdx.x * blockDim.x + threadIdx.x;
    if (t >= M * 3) return;
    const int c = t % 3, r = t / 3;
    float acc = bias[c];
    const float* hr = h2 + (size_t)r * 128;
#pragma unroll 4
    for (int k = 0; k < 128; ++k) acc += hr[k] * W[k * 3 + c];
    out[t] = u[t] + acc;
}

// ---------------------------------------------------------------------------
// Host orchestration
// ---------------------------------------------------------------------------
struct MLPd {
    const float *ln_s, *ln_b;
    const float *W0, *b0, *W1, *b1, *W2, *b2;
    bf16_t *W0p, *W1p, *W2p;
};

extern "C" void kernel_launch(void* const* d_in, const int* in_sizes, int n_in,
                              void* d_out, int out_size, void* d_ws, size_t ws_size,
                              hipStream_t stream)
{
    const float* u = (const float*)d_in[0];
    int p = 1;
    auto F = [&](int i) { return (const float*)d_in[i]; };
    auto take_ln = [&]() { MLPd m{}; m.ln_b = F(p++); m.ln_s = F(p++);
        m.W0 = F(p++); m.b0 = F(p++); m.W1 = F(p++); m.b1 = F(p++);
        m.W2 = F(p++); m.b2 = F(p++); return m; };
    auto take_noln = [&]() { MLPd m{}; m.ln_b = nullptr; m.ln_s = nullptr;
        m.W0 = F(p++); m.b0 = F(p++); m.W1 = F(p++); m.b1 = F(p++);
        m.W2 = F(p++); m.b2 = F(p++); return m; };

    // pytree order: sorted keys inside 'params'
    MLPd g2m_edge  = take_ln();   // g2m: edge, emb_edge, emb_grid, emb_mesh, node_grid, node_mesh
    MLPd g2m_embE  = take_ln();
    MLPd g2m_embG  = take_ln();
    MLPd g2m_embM  = take_ln();
    MLPd g2m_nodeG = take_ln();
    MLPd g2m_nodeM = take_ln();
    MLPd m2g_edge  = take_ln();   // m2g: edge, emb_edge, node_grid, out
    MLPd m2g_embE  = take_ln();
    MLPd m2g_nodeG = take_ln();
    MLPd m2g_out   = take_noln();
    MLPd mesh_embE = take_ln();   // mesh: emb_edge, steps[0..5]{edge,node}
    MLPd step_edge[6], step_node[6];
    for (int s = 0; s < 6; ++s) { step_edge[s] = take_ln(); step_node[s] = take_ln(); }

    const float* grid_sf = F(p); const long Ng = in_sizes[p] / 4; ++p;
    const float* mesh_sf = F(p); const long Nm = in_sizes[p] / 4; ++p;
    const float* g2m_ef  = F(p); const long E  = in_sizes[p] / 6; ++p;
    const float* mesh_ef = F(p); ++p;
    const float* m2g_ef  = F(p); ++p;
    const int* g2m_g = (const int*)d_in[p++];
    const int* g2m_m = (const int*)d_in[p++];
    const int* mm_s  = (const int*)d_in[p]; const long Emm = in_sizes[p]; ++p;
    const int* mm_r  = (const int*)d_in[p++];

    const size_t Mg  = (size_t)Ng * BB;
    const size_t Mm  = (size_t)Nm * BB;
    const size_t Me  = (size_t)E * BB;
    const size_t Mme = (size_t)Emm * BB;
    size_t maxR = Mg; if (Me > maxR) maxR = Me; if (Mme > maxR) maxR = Mme;

    // ---------------- workspace layout (fp32 + packed bf16 weights) --------
    float* wsf = (float*)d_ws;
    size_t o = 0;
    float* gbuf  = wsf + o; o += Mg  * 128;
    float* mbuf  = wsf + o; o += Mm  * 128;
    float* ebuf  = wsf + o; o += Me  * 128;   // g2m edges; reused for m2g edges
    float* embuf = wsf + o; o += Mme * 128;   // mesh edges
    float* h1    = wsf + o; o += maxR * 128;  // hidden 1 / pre-LN linear
    float* h2    = wsf + o; o += maxR * 128;  // hidden 2
    float* aggb  = wsf + o; o += Mg  * 128;   // segment-mean target (max Ng rows)
    float* cntb  = wsf + o; o += (size_t)Ng;  // counts (max Ng entities)
    bf16_t* wp = (bf16_t*)(wsf + o);          // packed-weight arena

    auto packW = [&](const float* W, int K) -> bf16_t* {
        bf16_t* dst = wp;
        const int total = (K >> 5) * 4096;
        wp += (size_t)total;
        pack_w_k<<<dim3((unsigned)((total + 255) / 256)), 256, 0, stream>>>(W, dst, K);
        return dst;
    };
    auto prep = [&](MLPd& m, int K0, bool packLast) {
        if (K0 >= 128) m.W0p = packW(m.W0, K0);
        m.W1p = packW(m.W1, 128);
        if (packLast) m.W2p = packW(m.W2, 128);
    };
    prep(g2m_edge, 384, true);  prep(g2m_embE, 6, true);  prep(g2m_embG, 7, true);
    prep(g2m_embM, 4, true);    prep(g2m_nodeG, 128, true); prep(g2m_nodeM, 256, true);
    prep(m2g_edge, 384, true);  prep(m2g_embE, 6, true);  prep(m2g_nodeG, 256, true);
    prep(m2g_out, 128, false);  // W2 is 128x3: consumed raw by out_final_k
    prep(mesh_embE, 6, true);
    for (int s = 0; s < 6; ++s) { prep(step_edge[s], 384, true); prep(step_node[s], 256, true); }

    // ---------------- launch helpers ----------------
    auto fast = [&](int nseg,
                    const float* a0, const int* i0, int m0,
                    const float* a1, const int* i1, int m1,
                    const float* a2, const int* i2, int m2,
                    const bf16_t* Wp, const float* bias, float* outp, size_t M, int act) {
        dim3 g((unsigned)((M + 15) / 16));
        if (act == 1) {
            if (nseg == 1)
                wmma_mlp_fast<1, 1><<<g, 256, 0, stream>>>(a0, i0, m0, a1, i1, m1, a2, i2, m2,
                                                           Wp, bias, outp, (int)M);
            else if (nseg == 2)
                wmma_mlp_fast<2, 1><<<g, 256, 0, stream>>>(a0, i0, m0, a1, i1, m1, a2, i2, m2,
                                                           Wp, bias, outp, (int)M);
            else
                wmma_mlp_fast<3, 1><<<g, 256, 0, stream>>>(a0, i0, m0, a1, i1, m1, a2, i2, m2,
                                                           Wp, bias, outp, (int)M);
        } else {
            if (nseg == 1)
                wmma_mlp_fast<1, 0><<<g, 256, 0, stream>>>(a0, i0, m0, a1, i1, m1, a2, i2, m2,
                                                           Wp, bias, outp, (int)M);
            else if (nseg == 2)
                wmma_mlp_fast<2, 0><<<g, 256, 0, stream>>>(a0, i0, m0, a1, i1, m1, a2, i2, m2,
                                                           Wp, bias, outp, (int)M);
            else
                wmma_mlp_fast<3, 0><<<g, 256, 0, stream>>>(a0, i0, m0, a1, i1, m1, a2, i2, m2,
                                                           Wp, bias, outp, (int)M);
        }
    };
    auto tail23 = [&](const MLPd& mp, const float* res, float* dst, size_t M) {
        fast(1, h1, nullptr, 0, nullptr, nullptr, 0, nullptr, nullptr, 0, mp.W1p, mp.b1, h2, M, 1);
        fast(1, h2, nullptr, 0, nullptr, nullptr, 0, nullptr, nullptr, 0, mp.W2p, mp.b2, h1, M, 0);
        ln_res_k<<<dim3((unsigned)((M * 32 + 255) / 256)), 256, 0, stream>>>(
            h1, mp.ln_s, mp.ln_b, res, dst, (int)M);
    };
    auto run_mlp_fast = [&](const MLPd& mp, int nseg,
                            const float* a0, const int* i0, int m0,
                            const float* a1, const int* i1, int m1,
                            const float* a2, const int* i2, int m2,
                            float* dst, const float* res, size_t M) {
        fast(nseg, a0, i0, m0, a1, i1, m1, a2, i2, m2, mp.W0p, mp.b0, h1, M, 1);
        tail23(mp, res, dst, M);
    };
    auto run_mlp_emb = [&](const MLPd& mp,
                           const float* a0, int d0, int md0,
                           const float* a1, int d1, int md1,
                           float* dst, size_t M) {
        wmma_mlp_generic<<<dim3((unsigned)((M + 15) / 16)), 256, 0, stream>>>(
            a0, d0, md0, a1, d1, md1, mp.W0, mp.b0, h1, (int)M, d0 + d1);
        tail23(mp, nullptr, dst, M);
    };
    auto segmean = [&](const float* feat, const int* idx, size_t Ecnt, size_t Nent) {
        hipMemsetAsync(aggb, 0, Nent * BB * 128 * sizeof(float), stream);
        hipMemsetAsync(cntb, 0, Nent * sizeof(float), stream);
        const long tot = (long)Ecnt * BB * 128;
        scatter_add_k<<<dim3((unsigned)((tot + 255) / 256)), 256, 0, stream>>>(feat, idx, aggb, tot);
        count_k<<<dim3((unsigned)((Ecnt + 255) / 256)), 256, 0, stream>>>(idx, cntb, (int)Ecnt);
        const long rows = (long)Nent * BB;
        divide_k<<<dim3((unsigned)((rows * 128 + 255) / 256)), 256, 0, stream>>>(aggb, cntb, rows);
    };

    // ---------------- encoder: grid2mesh ----------------
    run_mlp_emb(g2m_embG, u, 3, 0, grid_sf, 4, 2, gbuf, Mg);
    run_mlp_emb(g2m_embM, mesh_sf, 4, 2, nullptr, 0, 0, mbuf, Mm);
    run_mlp_emb(g2m_embE, g2m_ef, 6, 2, nullptr, 0, 0, ebuf, Me);
    run_mlp_fast(g2m_edge, 3, ebuf, nullptr, 0, gbuf, g2m_g, 1, mbuf, g2m_m, 1,
                 ebuf, ebuf, Me);
    segmean(ebuf, g2m_m, (size_t)E, (size_t)Nm);
    run_mlp_fast(g2m_nodeM, 2, mbuf, nullptr, 0, aggb, nullptr, 0, nullptr, nullptr, 0,
                 mbuf, mbuf, Mm);
    run_mlp_fast(g2m_nodeG, 1, gbuf, nullptr, 0, nullptr, nullptr, 0, nullptr, nullptr, 0,
                 gbuf, gbuf, Mg);

    // ---------------- processor: multimesh GNN ----------------
    run_mlp_emb(mesh_embE, mesh_ef, 6, 2, nullptr, 0, 0, embuf, Mme);
    for (int s = 0; s < 6; ++s) {
        run_mlp_fast(step_edge[s], 3, embuf, nullptr, 0, mbuf, mm_s, 1, mbuf, mm_r, 1,
                     embuf, embuf, Mme);
        segmean(embuf, mm_r, (size_t)Emm, (size_t)Nm);
        run_mlp_fast(step_node[s], 2, mbuf, nullptr, 0, aggb, nullptr, 0, nullptr, nullptr, 0,
                     mbuf, mbuf, Mm);
    }

    // ---------------- decoder: mesh2grid ----------------
    run_mlp_emb(m2g_embE, m2g_ef, 6, 2, nullptr, 0, 0, ebuf, Me);
    run_mlp_fast(m2g_edge, 3, ebuf, nullptr, 0, mbuf, g2m_m, 1, gbuf, g2m_g, 1,
                 ebuf, ebuf, Me);
    segmean(ebuf, g2m_g, (size_t)E, (size_t)Ng);
    run_mlp_fast(m2g_nodeG, 2, gbuf, nullptr, 0, aggb, nullptr, 0, nullptr, nullptr, 0,
                 gbuf, gbuf, Mg);

    // output MLP (no LN, dout=3) + residual update
    fast(1, gbuf, nullptr, 0, nullptr, nullptr, 0, nullptr, nullptr, 0,
         m2g_out.W0p, m2g_out.b0, h1, Mg, 1);
    fast(1, h1, nullptr, 0, nullptr, nullptr, 0, nullptr, nullptr, 0,
         m2g_out.W1p, m2g_out.b1, h2, Mg, 1);
    out_final_k<<<dim3((unsigned)((Mg * 3 + 255) / 256)), 256, 0, stream>>>(
        h2, m2g_out.W2, m2g_out.b2, u, (float*)d_out, (int)Mg);
}
